// SpatialAttentionLayer_36120674959406
// MI455X (gfx1250) — matrix operand order repaired and checked by reading, not compiled, not run on hardware
//
#include <hip/hip_runtime.h>
#include <hip/hip_bf16.h>

// ---------------------------------------------------------------------------
// SAGAN spatial attention, fused flash-attention style, CDNA5 WMMA (gfx1250).
//   B=4, N=64*64=4096, C=256, DQK=32.
//   ~41 GFLOP dominated by P@V -> v_wmma_f32_16x16x32_f16.
//   K/V tiles staged into LDS with CDNA5 async global->LDS copies
//   (ASYNCcnt / s_wait_asynccnt), double-buffered, shared by all 8 waves of
//   a workgroup -> 8x less L2 traffic and fetch overlaps the WMMA stream.
//   Round 3: LDS rows padded 32->40 halves (80B stride, bank stride 20) so
//   the per-lane ds_load_b128 windows hit each 4-bank group exactly twice
//   (the 512B/instr floor) instead of 4-way conflicting at 64B stride.
// ---------------------------------------------------------------------------

typedef __attribute__((ext_vector_type(16))) _Float16 v16h;
typedef __attribute__((ext_vector_type(8)))  _Float16 v8h;
typedef __attribute__((ext_vector_type(8)))  float    v8f;
typedef __attribute__((ext_vector_type(4)))  float    v4f;

union V16H { v16h v; v8h h[2]; };

#define WMMA_F16(A, Bm, Cm) \
  __builtin_amdgcn_wmma_f32_16x16x32_f16(false, (A), false, (Bm), (short)0, (Cm), false, false)

__device__ __forceinline__ float xor16(float x) { return __shfl_xor(x, 16, 32); }

// CDNA5 async global->LDS copy, 16B per lane, tracked by ASYNCcnt.
__device__ __forceinline__ void async_b128(unsigned lds_off, const void* g) {
  asm volatile("global_load_async_to_lds_b128 %0, %1, off"
               :: "v"(lds_off), "v"(g) : "memory");
}
__device__ __forceinline__ void wait_async0() {
  asm volatile("s_wait_asynccnt 0" ::: "memory");
}

// dims
#define NTOK   4096
#define CDIM   256
#define DQK    32
#define NBATCH 4
#define LPAD   40   // padded LDS row stride in halves (80B -> bank stride 20)

// ---------------------------------------------------------------------------
// Kernel 0: transpose + f16-convert weights so projection B-tiles are
// contiguous 32-byte loads. Wf/Wg: [256][32] -> [32][256]; Wh: [256][256]^T.
// ---------------------------------------------------------------------------
__global__ void wt_kernel(const float* __restrict__ Wf, const float* __restrict__ Wg,
                          const float* __restrict__ Wh,
                          _Float16* __restrict__ WfT, _Float16* __restrict__ WgT,
                          _Float16* __restrict__ WhT) {
  int i = blockIdx.x * blockDim.x + threadIdx.x;
  if (i < DQK * CDIM) {
    int d = i / CDIM, c = i % CDIM;
    WfT[i] = (_Float16)Wf[c * DQK + d];
    WgT[i] = (_Float16)Wg[c * DQK + d];
  }
  for (int j = i; j < CDIM * CDIM; j += gridDim.x * blockDim.x) {
    int co = j / CDIM, ci = j % CDIM;
    WhT[j] = (_Float16)Wh[ci * CDIM + co];
  }
}

// ---------------------------------------------------------------------------
// Kernel 1: projections via WMMA.
//   One wave = 16 token rows. K-loop over C in chunks of 32.
//   Outputs: Qh, Kh row-major [16384][32] f16; Vt per-batch [C][N] f16.
// ---------------------------------------------------------------------------
__global__ __launch_bounds__(256) void proj_kernel(
    const float* __restrict__ x,
    const _Float16* __restrict__ WfT, const _Float16* __restrict__ WgT,
    const _Float16* __restrict__ WhT,
    const float* __restrict__ bf, const float* __restrict__ bg,
    const float* __restrict__ bh,
    _Float16* __restrict__ Qh, _Float16* __restrict__ Kh,
    _Float16* __restrict__ Vt) {
  const int wave = threadIdx.x >> 5;
  const int lane = threadIdx.x & 31;
  const int ln = lane & 15, lh = lane >> 4;
  const int t    = blockIdx.x * 8 + wave;   // 0..1023 (16-row tile of 16384 tokens)
  const int row0 = t * 16;

  v8f zero = {};
  v8f accq0 = zero, accq1 = zero, acck0 = zero, acck1 = zero;
  v8f accv[16];
#pragma unroll
  for (int i = 0; i < 16; ++i) accv[i] = zero;

  const float* xrow = x + (size_t)(row0 + ln) * CDIM;

  for (int kc = 0; kc < CDIM; kc += 32) {
    // A tile (16 tokens x 32 channels), f32 -> f16, contiguous-k A-layout (same as B):
    // lane holds K = lh*16 + [0..15] of row ln.
    V16H a;
    {
      v4f x0 = *(const v4f*)(xrow + kc + lh * 16);
      v4f x1 = *(const v4f*)(xrow + kc + lh * 16 + 4);
      v4f x2 = *(const v4f*)(xrow + kc + lh * 16 + 8);
      v4f x3 = *(const v4f*)(xrow + kc + lh * 16 + 12);
#pragma unroll
      for (int j = 0; j < 4; ++j) {
        a.v[j]      = (_Float16)x0[j];
        a.v[4 + j]  = (_Float16)x1[j];
        a.v[8 + j]  = (_Float16)x2[j];
        a.v[12 + j] = (_Float16)x3[j];
      }
    }
    // B tiles: lane holds col n = tile+ln, K = kc + lh*16 + [0..15] (contiguous)
    accq0 = WMMA_F16(a.v, *(const v16h*)(WgT + (size_t)(ln)      * CDIM + kc + lh * 16), accq0);
    accq1 = WMMA_F16(a.v, *(const v16h*)(WgT + (size_t)(16 + ln) * CDIM + kc + lh * 16), accq1);
    acck0 = WMMA_F16(a.v, *(const v16h*)(WfT + (size_t)(ln)      * CDIM + kc + lh * 16), acck0);
    acck1 = WMMA_F16(a.v, *(const v16h*)(WfT + (size_t)(16 + ln) * CDIM + kc + lh * 16), acck1);
#pragma unroll
    for (int ct = 0; ct < 16; ++ct) {
      const v16h bw = *(const v16h*)(WhT + (size_t)(ct * 16 + ln) * CDIM + kc + lh * 16);
      accv[ct] = WMMA_F16(a.v, bw, accv[ct]);
    }
  }

  // D-layout: reg r, lane -> element [M = r + 8*lh][N = ln]
  const float bq0 = bg[ln], bq1 = bg[16 + ln];
  const float bk0 = bf[ln], bk1 = bf[16 + ln];
#pragma unroll
  for (int r = 0; r < 8; ++r) {
    size_t rowm = (size_t)(row0 + r + 8 * lh);
    Qh[rowm * DQK + ln]      = (_Float16)(accq0[r] + bq0);
    Qh[rowm * DQK + 16 + ln] = (_Float16)(accq1[r] + bq1);
    Kh[rowm * DQK + ln]      = (_Float16)(acck0[r] + bk0);
    Kh[rowm * DQK + 16 + ln] = (_Float16)(acck1[r] + bk1);
  }
  // V stored transposed per batch: Vt[b][c][n_local]; tokens are contiguous
  // along r -> one 16-byte store per c-tile per lane.
  const int b  = t >> 8;                  // t / 256
  const int tl = (t & 255) * 16 + 8 * lh; // token_local base for this half
#pragma unroll
  for (int ct = 0; ct < 16; ++ct) {
    int c = ct * 16 + ln;
    float bb = bh[c];
    v8h pk;
#pragma unroll
    for (int r = 0; r < 8; ++r) pk[r] = (_Float16)(accv[ct][r] + bb);
    *(v8h*)(Vt + ((size_t)(b * CDIM + c)) * NTOK + tl) = pk;
  }
}

// ---------------------------------------------------------------------------
// Kernel 2: fused flash attention + residual, LDS double-buffered K/V.
//   Block = 8 waves, all in the same batch -> shared K/V stream.
//   Per 32-key block (staged in LDS): 2 WMMAs (S^T), online softmax
//   (per-lane + shfl_xor16), 16 WMMAs (O^T += V^T * P).
//   Epilogue: out = gamma*O/l + x.
// ---------------------------------------------------------------------------
__global__ __launch_bounds__(256) void attn_kernel(
    const float* __restrict__ x,
    const _Float16* __restrict__ Qh, const _Float16* __restrict__ Kh,
    const _Float16* __restrict__ Vt,
    const float* __restrict__ gamma_p,
    float* __restrict__ out) {
  __shared__ __attribute__((aligned(32))) _Float16 sK[2][32 * LPAD];    // 2 x 2.5KB
  __shared__ __attribute__((aligned(32))) _Float16 sV[2][CDIM * LPAD];  // 2 x 20KB

  const int tid  = threadIdx.x;
  const int wave = tid >> 5;
  const int lane = tid & 31;
  const int ln = lane & 15, lh = lane >> 4;
  const int t  = blockIdx.x * 8 + wave;   // 0..1023
  const int b  = t >> 8;                  // whole block shares one batch
  const int qt = t & 255;
  const int q  = b * NTOK + qt * 16 + ln; // this lane's query token (global)

  const _Float16* Kb = Kh + (size_t)b * NTOK * DQK;
  const _Float16* Vb = Vt + (size_t)b * CDIM * NTOK;

  // Q^T as B-matrix: lane holds d = lh*16 + [0..15] of query q (contiguous)
  const v16h bq = *(const v16h*)(Qh + (size_t)q * DQK + lh * 16);

  v8f zero = {};
  v8f acc[16];
#pragma unroll
  for (int i = 0; i < 16; ++i) acc[i] = zero;
  float m = -__builtin_inff();
  float l = 0.f;

  // cooperative async stage of one 32-key block into LDS buffer `buf`
  auto stage = [&](int buf, int kt) {
    // K tile: 32 rows x 64B payload = 128 x 16B chunks (threads 0..127)
    if (tid < 128) {
      int row = tid >> 2, off = (tid & 3) * 8;  // off in halves
      async_b128((unsigned)(uintptr_t)(&sK[buf][row * LPAD + off]),
                 Kb + (size_t)(kt + row) * DQK + off);
    }
    // V tile: 256 rows x 64B payload = 1024 x 16B chunks (4 per thread)
#pragma unroll
    for (int u = 0; u < 4; ++u) {
      int c = tid * 4 + u;
      int row = c >> 2, off = (c & 3) * 8;
      async_b128((unsigned)(uintptr_t)(&sV[buf][row * LPAD + off]),
                 Vb + (size_t)row * NTOK + kt + off);
    }
  };

  stage(0, 0);

  for (int it = 0; it < NTOK / 32; ++it) {
    const int cur = it & 1;
    wait_async0();          // this wave's async fills of sK/sV[cur] done
    __syncthreads();        // everyone's fills done, everyone past prior reads
    if (it + 1 < NTOK / 32) stage(cur ^ 1, (it + 1) * 32);  // overlap next fetch

    // Key A-tiles from LDS: rows = key tokens, K = head dim
    V16H ak0, ak1;
    {
      const _Float16* k0 = &sK[cur][(size_t)ln * LPAD];
      const _Float16* k1 = &sK[cur][(size_t)(16 + ln) * LPAD];
      ak0.h[0] = *(const v8h*)(k0 + lh * 16);
      ak0.h[1] = *(const v8h*)(k0 + lh * 16 + 8);
      ak1.h[0] = *(const v8h*)(k1 + lh * 16);
      ak1.h[1] = *(const v8h*)(k1 + lh * 16 + 8);
    }
    // S^T tiles: [key][query]; query pinned to lane%16
    v8f sl = WMMA_F16(ak0.v, bq, zero);   // keys kt+0..15
    v8f sh = WMMA_F16(ak1.v, bq, zero);   // keys kt+16..31

    // online softmax over keys for this lane's query
    float bm = sl[0];
#pragma unroll
    for (int r = 1; r < 8; ++r) bm = fmaxf(bm, sl[r]);
#pragma unroll
    for (int r = 0; r < 8; ++r) bm = fmaxf(bm, sh[r]);
    bm = fmaxf(bm, xor16(bm));
    float mn   = fmaxf(m, bm);
    float corr = __expf(m - mn);
    float rs = 0.f;
#pragma unroll
    for (int r = 0; r < 8; ++r) { sl[r] = __expf(sl[r] - mn); rs += sl[r]; }
#pragma unroll
    for (int r = 0; r < 8; ++r) { sh[r] = __expf(sh[r] - mn); rs += sh[r]; }
    rs += xor16(rs);
    l = l * corr + rs;
    m = mn;
#pragma unroll
    for (int i = 0; i < 16; ++i)
#pragma unroll
      for (int r = 0; r < 8; ++r) acc[i][r] *= corr;

    // Reshape exp(S^T) (D-layout) into B-matrix P (32 keys x 16 queries):
    // lane needs k = lh*16 + j; cross-half values via shfl_xor(16).
    V16H bp;
#pragma unroll
    for (int j = 0; j < 8; ++j) {
      float lox = xor16(sl[j]);
      float hix = xor16(sh[j]);
      bp.v[j]     = (_Float16)((lh == 0) ? sl[j] : hix);
      bp.v[j + 8] = (_Float16)((lh == 0) ? lox : sh[j]);
    }

    // O^T += V^T * P across 16 c-tiles (V^T tiles from LDS)
#pragma unroll
    for (int ct = 0; ct < 16; ++ct) {
      const _Float16* vr = &sV[cur][(size_t)(ct * 16 + ln) * LPAD + lh * 16];
      V16H av;
      av.h[0] = *(const v8h*)(vr);
      av.h[1] = *(const v8h*)(vr + 8);
      acc[ct] = WMMA_F16(av.v, bp.v, acc[ct]);
    }

    __syncthreads();        // all waves done reading buffer `cur`
  }

  // epilogue: out[q][c] = gamma * O[q][c] / l + x[q][c]
  const float g    = gamma_p[0];
  const float linv = g / l;
  const size_t obase = (size_t)q * CDIM;
#pragma unroll
  for (int ct = 0; ct < 16; ++ct) {
    size_t o = obase + (size_t)(ct * 16 + 8 * lh);
    v4f x0 = *(const v4f*)(x + o);
    v4f x1 = *(const v4f*)(x + o + 4);
    v4f r0, r1;
#pragma unroll
    for (int r = 0; r < 4; ++r) {
      r0[r] = acc[ct][r] * linv + x0[r];
      r1[r] = acc[ct][4 + r] * linv + x1[r];
    }
    *(v4f*)(out + o) = r0;
    *(v4f*)(out + o + 4) = r1;
  }
}

// ---------------------------------------------------------------------------
extern "C" void kernel_launch(void* const* d_in, const int* in_sizes, int n_in,
                              void* d_out, int out_size, void* d_ws, size_t ws_size,
                              hipStream_t stream) {
  const float* x     = (const float*)d_in[0];
  const float* Wf    = (const float*)d_in[1];
  const float* bf    = (const float*)d_in[2];
  const float* Wg    = (const float*)d_in[3];
  const float* bg    = (const float*)d_in[4];
  const float* Wh    = (const float*)d_in[5];
  const float* bh    = (const float*)d_in[6];
  const float* gamma = (const float*)d_in[7];
  float* out = (float*)d_out;

  // workspace layout (halves); all region starts 256B-aligned
  _Float16* ws  = (_Float16*)d_ws;
  _Float16* WfT = ws;                         //  32*256   =    8192
  _Float16* WgT = WfT + 8192;                 //  32*256   =    8192
  _Float16* WhT = WgT + 8192;                 // 256*256   =   65536
  _Float16* Qh  = WhT + 65536;                // 16384*32  =  524288
  _Float16* Kh  = Qh + 524288;                // 16384*32  =  524288
  _Float16* Vt  = Kh + 524288;                // 4*256*4096= 4194304
  (void)ws_size; (void)n_in; (void)in_sizes; (void)out_size;

  wt_kernel<<<256, 256, 0, stream>>>(Wf, Wg, Wh, WfT, WgT, WhT);
  proj_kernel<<<128, 256, 0, stream>>>(x, WfT, WgT, WhT, bf, bg, bh, Qh, Kh, Vt);
  attn_kernel<<<128, 256, 0, stream>>>(x, Qh, Kh, Vt, gamma, out);
}
